// RNN_59992103190962
// MI455X (gfx1250) — compile-verified, matrix operand order checked by reading, not run on hardware
//
#include <hip/hip_runtime.h>

typedef __attribute__((ext_vector_type(16))) __bf16 v16bf;
typedef __attribute__((ext_vector_type(8)))  float  v8f;

#define NWG    32
#define HDIM   512
#define BDIM   64
#define TSTEPS 512
#define CHK    64          // K-chunk staged to LDS per double-buffer slot

// ---------------------------------------------------------------------------
// Fragment loader. Per CDNA5 ISA 7.12.2 (16-bit A-matrix 16x32):
//   lanes 0-15 (half=0): elems 0..7 -> K=kk+0..7,  elems 8..15 -> K=kk+16..23
//   lanes 16-31(half=1): elems 0..7 -> K=kk+8..15, elems 8..15 -> K=kk+24..31
// Two contiguous 16-byte chunks per lane -> global_load_b128 / ds_load_b128.
// ---------------------------------------------------------------------------
__device__ __forceinline__ v16bf load_frag16(const __bf16* rowp, int kk, int half) {
    v16bf f;
    ((float4*)&f)[0] = *reinterpret_cast<const float4*>(rowp + kk + half * 8);
    ((float4*)&f)[1] = *reinterpret_cast<const float4*>(rowp + kk + 16 + half * 8);
    return f;
}

__device__ __forceinline__ float sigf(float x) {
    return 1.0f / (1.0f + __expf(-x));
}

// Atomic grid-wide barrier over NWG persistent workgroups (monotonic counter).
__device__ __forceinline__ void grid_barrier(unsigned* ctr) {
    __threadfence();
    __syncthreads();
    if (threadIdx.x == 0) {
        unsigned arrived = atomicAdd(ctr, 1u) + 1u;
        unsigned target  = ((arrived + NWG - 1u) / NWG) * NWG;
        while (__hip_atomic_load(ctr, __ATOMIC_ACQUIRE, __HIP_MEMORY_SCOPE_AGENT) < target) {
            __builtin_amdgcn_s_sleep(2);
        }
    }
    __syncthreads();
}

// ---------------------------------------------------------------------------
// Persistent LSTM layer scan. Grid = NWG x 256 threads (8 waves, wave32).
// WG owns hidden units [16*wg, 16*wg+16); wave = (gate, M-half).
// gates = [x_t ; h_{t-1}] (64 x Ktot) · Wcat^T slice, bf16 WMMA, f32 acc.
// A matrix is staged into LDS in 64-wide K chunks via the gfx1250 async
// copy engine (global_load_async_to_lds_b128, ASYNCcnt), double-buffered.
// ---------------------------------------------------------------------------
__global__ __launch_bounds__(256, 1) void k_lstm_scan(
    const __bf16* __restrict__ xin, int Kin,        // [T, B, Kin] bf16
    __bf16* __restrict__ yout,                      // [T, B, 512] bf16
    const __bf16* __restrict__ Wcat,                // [2048, Kin+512] bf16
    const float*  __restrict__ bias,                // [2048] (b_ih + b_hh)
    __bf16* __restrict__ hbuf,                      // [2, 64, 512] ping-pong
    unsigned* __restrict__ ctr,
    float* __restrict__ hlast,                      // [64,512] f32 or null
    int T)
{
    __shared__ __bf16 abuf[2][BDIM * CHK];          // 2 x 8KB A staging
    __shared__ float  gbuf[4][BDIM][17];            // gates for elementwise

    const int Ktot = Kin + HDIM;
    const int tid  = threadIdx.x;
    const int wave = tid >> 5;
    const int lane = tid & 31;
    const int half = lane >> 4;
    const int l16  = lane & 15;
    const int gate = wave & 3;                      // i, f, g, o
    const int mh   = wave >> 2;                     // batch half (rows 32*mh..)
    const int hbase = blockIdx.x * (HDIM / NWG);    // 16-unit hidden slice
    const int ncol  = gate * HDIM + hbase + l16;    // this lane's gate column
    const __bf16* wrow = Wcat + (size_t)ncol * Ktot;
    const float bv = bias[ncol];

    // Per-thread async-copy slots: chunk = 64 rows x 64 bf16 cols = 512 B128
    // transfers; 256 threads x 2 slots. tr -> row = tr>>3, seg = tr&7.
    size_t  xoff_q[2], hoff_q[2];
    unsigned ldsoff_q[2];
#pragma unroll
    for (int q = 0; q < 2; ++q) {
        int tr  = tid + q * 256;
        int row = tr >> 3;
        int seg = tr & 7;
        xoff_q[q]   = (size_t)row * Kin  + seg * 8;
        hoff_q[q]   = (size_t)row * HDIM + seg * 8;
        ldsoff_q[q] = (unsigned)(size_t)&abuf[0][row * CHK + seg * 8];
    }
    const unsigned abstride = (unsigned)(BDIM * CHK * sizeof(__bf16)); // 8192

    float creg[4] = {0.f, 0.f, 0.f, 0.f};          // cell state in registers

    for (int t = 0; t < T; ++t) {
        const __bf16* xt    = xin  + (size_t)t * BDIM * Kin;
        const __bf16* hprev = hbuf + (size_t)(t & 1) * BDIM * HDIM;
        __bf16*       hcur  = hbuf + (size_t)((t + 1) & 1) * BDIM * HDIM;

        const int kend = (t == 0) ? Kin : Ktot;     // h_0 == 0: skip h columns
        const int nc   = kend / CHK;

        auto issue_chunk = [&](int c) {
            const int kg = c * CHK;
            const unsigned dbase = (c & 1) ? abstride : 0u;
#pragma unroll
            for (int q = 0; q < 2; ++q) {
                const __bf16* src = (kg < Kin)
                    ? (xt    + xoff_q[q] + kg)
                    : (hprev + hoff_q[q] + (kg - Kin));
                unsigned           ldsa = ldsoff_q[q] + dbase;
                unsigned long long ga   = (unsigned long long)(size_t)src;
                asm volatile("global_load_async_to_lds_b128 %0, %1, off"
                             :: "v"(ldsa), "v"(ga) : "memory");
            }
        };

        v8f acc[2] = {};
        issue_chunk(0);
        for (int c = 0; c < nc; ++c) {
            if (c + 1 < nc) {
                issue_chunk(c + 1);                         // prefetch next chunk
                asm volatile("s_wait_asynccnt 0x2" ::: "memory");
            } else {
                asm volatile("s_wait_asynccnt 0x0" ::: "memory");
            }
            __syncthreads();                                // chunk c visible WG-wide

            const __bf16* ab = &abuf[c & 1][0];
            const int kg = c * CHK;
            __builtin_prefetch(wrow + kg + 2 * CHK, 0, 1);  // global_prefetch_b8
#pragma unroll
            for (int kk2 = 0; kk2 < CHK; kk2 += 32) {
                v16bf bfrag = load_frag16(wrow, kg + kk2, half);
#pragma unroll
                for (int mt = 0; mt < 2; ++mt) {
                    int r = 32 * mh + 16 * mt + l16;
                    v16bf afrag = load_frag16(ab + r * CHK, kk2, half);
                    acc[mt] = __builtin_amdgcn_wmma_f32_16x16x32_bf16(
                        false, afrag, false, bfrag, (short)0, acc[mt], false, false);
                }
            }
            __syncthreads();                                // done reading buf c
        }

        // Stage gate pre-activations to LDS (C layout: M = i + 8*half, 7.12.2)
#pragma unroll
        for (int mt = 0; mt < 2; ++mt)
#pragma unroll
            for (int i = 0; i < 8; ++i) {
                int b = 32 * mh + 16 * mt + i + 8 * half;
                gbuf[gate][b][l16] = acc[mt][i] + bv;
            }
        __syncthreads();

        // Elementwise LSTM cell: 64 batch x 16 units = 1024 elems / 256 threads
#pragma unroll
        for (int e = 0; e < 4; ++e) {
            int flat = tid * 4 + e;
            int b = flat >> 4;
            int j = flat & 15;
            float ig = sigf(gbuf[0][b][j]);
            float fg = sigf(gbuf[1][b][j]);
            float gg = tanhf(gbuf[2][b][j]);
            float og = sigf(gbuf[3][b][j]);
            float cc = fg * creg[e] + ig * gg;
            creg[e] = cc;
            float hh = og * tanhf(cc);
            int hidx = b * HDIM + hbase + j;
            hcur[hidx] = (__bf16)hh;
            yout[((size_t)t * BDIM + b) * HDIM + hbase + j] = (__bf16)hh;
            if (hlast != nullptr && t == T - 1) hlast[hidx] = hh;
        }
        grid_barrier(ctr);                          // h visible to all WGs for t+1
    }
}

// ---------------------------------------------------------------------------
// Prep kernels
// ---------------------------------------------------------------------------
__global__ void k_init(unsigned* ctr) {
    if (threadIdx.x == 0 && blockIdx.x == 0) *ctr = 0u;
}

// x [B=64, T=512, F=128] f32  ->  y [T, B, F] bf16
__global__ void k_to_bf16_tbf(const float* __restrict__ x, __bf16* __restrict__ y) {
    int idx = blockIdx.x * blockDim.x + threadIdx.x;
    if (idx >= 64 * 512 * 128) return;
    int f = idx & 127;
    int t = (idx >> 7) & 511;
    int b = idx >> 16;
    y[((size_t)t * 64 + b) * 128 + f] = (__bf16)x[idx];
}

// Build Wcat[n][k] = k<Kin ? W_ih[n][k] : W_hh[n][k-Kin] (bf16); bias = b_ih+b_hh
__global__ void k_prep(const float* __restrict__ Wih, const float* __restrict__ Whh,
                       const float* __restrict__ bih, const float* __restrict__ bhh,
                       int Kin, __bf16* __restrict__ Wcat, float* __restrict__ bias) {
    int Ktot = Kin + HDIM;
    int total = 4 * HDIM * Ktot;
    for (int idx = blockIdx.x * blockDim.x + threadIdx.x; idx < total;
         idx += gridDim.x * blockDim.x) {
        int n = idx / Ktot;
        int k = idx - n * Ktot;
        float v = (k < Kin) ? Wih[(size_t)n * Kin + k] : Whh[(size_t)n * HDIM + (k - Kin)];
        Wcat[idx] = (__bf16)v;
        if (idx < 4 * HDIM) bias[idx] = bih[idx] + bhh[idx];
    }
}

// FC head: relu(relu(h·W1^T + b1)·W2^T + b2). Tiny: 64 blocks x 64 threads.
__global__ __launch_bounds__(64) void k_fc(const float* __restrict__ h,
                                           const float* __restrict__ W1,
                                           const float* __restrict__ b1,
                                           const float* __restrict__ W2,
                                           const float* __restrict__ b2,
                                           float* __restrict__ out) {
    __shared__ float h1[64];
    int b = blockIdx.x, j = threadIdx.x;
    const float* hr = h + b * HDIM;
    const float* wr = W1 + j * HDIM;
    float s = b1[j];
    for (int k = 0; k < HDIM; ++k) s = fmaf(hr[k], wr[k], s);
    h1[j] = fmaxf(s, 0.f);
    __syncthreads();
    if (j == 0) {
        float s2 = b2[0];
        for (int k = 0; k < 64; ++k) s2 = fmaf(h1[k], W2[k], s2);
        out[b] = fmaxf(s2, 0.f);
    }
}

// ---------------------------------------------------------------------------
extern "C" void kernel_launch(void* const* d_in, const int* in_sizes, int n_in,
                              void* d_out, int out_size, void* d_ws, size_t ws_size,
                              hipStream_t stream) {
    const float* x     = (const float*)d_in[0];
    const float* Wih[3] = {(const float*)d_in[1], (const float*)d_in[5], (const float*)d_in[9]};
    const float* Whh[3] = {(const float*)d_in[2], (const float*)d_in[6], (const float*)d_in[10]};
    const float* bih[3] = {(const float*)d_in[3], (const float*)d_in[7], (const float*)d_in[11]};
    const float* bhh[3] = {(const float*)d_in[4], (const float*)d_in[8], (const float*)d_in[12]};
    const float* fcW1 = (const float*)d_in[13];
    const float* fcb1 = (const float*)d_in[14];
    const float* fcW2 = (const float*)d_in[15];
    const float* fcb2 = (const float*)d_in[16];
    float* out = (float*)d_out;

    char* p = (char*)d_ws;
    auto take = [&](size_t bytes) -> char* {
        char* r = p;
        p += (bytes + 255) & ~(size_t)255;
        return r;
    };
    unsigned* ctr   = (unsigned*)take(256);
    __bf16*   hbuf  = (__bf16*)take((size_t)2 * BDIM * HDIM * 2);
    float*    hlast = (float*)take((size_t)BDIM * HDIM * 4);
    float*  bias[3];
    __bf16* Wcat[3];
    const int Kin[3] = {128, HDIM, HDIM};
    for (int l = 0; l < 3; ++l) {
        bias[l] = (float*)take((size_t)4 * HDIM * 4);
        Wcat[l] = (__bf16*)take((size_t)4 * HDIM * (Kin[l] + HDIM) * 2);
    }
    __bf16* x0 = (__bf16*)take((size_t)TSTEPS * BDIM * 128 * 2);
    __bf16* y1 = (__bf16*)take((size_t)TSTEPS * BDIM * HDIM * 2);
    __bf16* y2 = (__bf16*)take((size_t)TSTEPS * BDIM * HDIM * 2);

    k_init<<<1, 32, 0, stream>>>(ctr);
    k_to_bf16_tbf<<<(64 * 512 * 128) / 256, 256, 0, stream>>>(x, x0);
    for (int l = 0; l < 3; ++l)
        k_prep<<<2048, 256, 0, stream>>>(Wih[l], Whh[l], bih[l], bhh[l], Kin[l], Wcat[l], bias[l]);

    k_lstm_scan<<<NWG, 256, 0, stream>>>(x0, 128, y1, Wcat[0], bias[0], hbuf, ctr, nullptr, TSTEPS);
    k_lstm_scan<<<NWG, 256, 0, stream>>>(y1, HDIM, y2, Wcat[1], bias[1], hbuf, ctr, nullptr, TSTEPS);
    k_lstm_scan<<<NWG, 256, 0, stream>>>(y2, HDIM, y1, Wcat[2], bias[2], hbuf, ctr, hlast, TSTEPS);

    k_fc<<<64, 64, 0, stream>>>(hlast, fcW1, fcb1, fcW2, fcb2, out);

    (void)in_sizes; (void)n_in; (void)out_size; (void)ws_size;
}
